// DualChunkAttention_35862976921925
// MI455X (gfx1250) — compile-verified, matrix operand order checked by reading, not run on hardware
//
#include <hip/hip_runtime.h>

// ---------------------------------------------------------------------------
// DualChunkAttention for MI455X (gfx1250, wave32, WMMA, async-LDS copies).
// combined_n = (sum_m w_m * softmax(Q_n K_m^T / sqrt(E))) @ V_n, w_m=2 iff m==n
// Compute-bound (~283 GFLOP vs ~150MB traffic) -> everything runs through
// v_wmma_f32_16x16x32_bf16; pure global->LDS tile copies use the CDNA5
// GLOBAL_LOAD_ASYNC_TO_LDS_B128 path (ASYNCcnt) with double-buffered LDS.
// ---------------------------------------------------------------------------

typedef __bf16 bf16;
typedef __attribute__((ext_vector_type(16))) __bf16 v16bf;
typedef __attribute__((ext_vector_type(8)))  __bf16 v8bf;
typedef __attribute__((ext_vector_type(8)))  float  v8f;

union V16 { v16bf v; bf16 e[16]; };
union V8F { v8f  v; float e[8]; };

#define DIM_B  4
#define DIM_S  4096
#define DIM_E  1024
#define DIM_C  256
#define DIM_NC 16

// ---- CDNA5 async global->LDS copy (B128), with portable fallback ----------
#if __has_builtin(__builtin_amdgcn_global_load_async_to_lds_b128) && \
    __has_builtin(__builtin_amdgcn_s_wait_asynccnt)
#define USE_ASYNC_COPY 1
// Builtin signature (from hipcc diagnostic): first param is
// 'int __attribute__((vector_size(16))) __device__ *', i.e. v4i in AS(1);
// second is the AS(3) analogue.
typedef int v4i_t __attribute__((vector_size(16)));
typedef __attribute__((address_space(1))) v4i_t as1_v4i;
typedef __attribute__((address_space(3))) v4i_t as3_v4i;
#else
#define USE_ASYNC_COPY 0
#endif

// Copy 16 bytes (8 bf16) global -> LDS.
__device__ __forceinline__ void copy16_g2l(const bf16* g, bf16* l) {
#if USE_ASYNC_COPY
    // AS(3) pointers are 32-bit; flat-addr low 32 bits are the LDS offset.
    __builtin_amdgcn_global_load_async_to_lds_b128(
        (as1_v4i*)(unsigned long long)g,
        (as3_v4i*)(unsigned int)(unsigned long long)l,
        /*offset=*/0, /*cpol=*/0);
#else
    *(v8bf*)l = *(const v8bf*)g;
#endif
}

__device__ __forceinline__ void wait_async_copies() {
#if USE_ASYNC_COPY
    __builtin_amdgcn_s_wait_asynccnt(0);
#endif
}

__device__ __forceinline__ v8f wmma_bf16(v16bf a, v16bf b, v8f c) {
    // D = A(16x32 bf16) * B(32x16 bf16) + C(16x16 f32)
    return __builtin_amdgcn_wmma_f32_16x16x32_bf16(
        /*neg_a=*/false, a, /*neg_b=*/false, b,
        /*c_mod=*/(short)0, c, /*reuse_a=*/false, /*reuse_b=*/false);
}

// ---------------------------------------------------------------------------
// fp32 -> bf16 conversion (one pass over x; enables raw async A-tile copies
// and halves the QKV A-stream footprint).
// ---------------------------------------------------------------------------
__global__ __launch_bounds__(256)
void cvt_f32_bf16_kernel(const float* __restrict__ src, bf16* __restrict__ dst) {
    const size_t i = ((size_t)blockIdx.x * 256 + threadIdx.x) * 8;
    #pragma unroll
    for (int j = 0; j < 8; ++j) dst[i + j] = (bf16)src[i + j];
}

// ---------------------------------------------------------------------------
// Tiled GEMM with bias: C[M,N] = A[M,K](bf16) @ W[K,N](f32) + bias[N]
// Block tile 128x128, 256 threads (8 waves, 2x4 wave grid, 64x32 per wave).
// A tile: raw async copy to LDS. W tile: fp32->bf16 + transpose while staging
// (Bt[n][k] so B-fragments are k-contiguous). Double-buffered, 1 barrier/step.
// ---------------------------------------------------------------------------
template <typename TOut>
__global__ __launch_bounds__(256)
void gemm_bias_kernel(const bf16* __restrict__ A, const float* __restrict__ W,
                      const float* __restrict__ bias, TOut* __restrict__ Cout,
                      int M, int N, int K) {
    __shared__ __align__(32) bf16 At[2][128][32];   // 16 KB
    __shared__ __align__(32) bf16 Bt[2][128][32];   // 16 KB

    const int tid  = threadIdx.x;
    const int lane = tid & 31;
    const int w    = tid >> 5;
    const int wm   = w >> 2;            // 0..1 (64 rows each)
    const int wn   = w & 3;             // 0..3 (32 cols each)
    const int rowBase = blockIdx.y * 128;
    const int colBase = blockIdx.x * 128;

    V8F acc[4][2];
    #pragma unroll
    for (int mt = 0; mt < 4; ++mt)
        #pragma unroll
        for (int nt = 0; nt < 2; ++nt)
            #pragma unroll
            for (int r = 0; r < 8; ++r) acc[mt][nt].e[r] = 0.0f;

    // A tile 128x32 bf16 = 512 16B-chunks; 2 chunks/thread, async.
    auto stageA = [&](int buf, int kk) {
        #pragma unroll
        for (int i = 0; i < 2; ++i) {
            const int chunk = tid * 2 + i;          // 0..511
            const int row = chunk >> 2, o8 = (chunk & 3) * 8;
            copy16_g2l(A + (size_t)(rowBase + row) * K + kk + o8, &At[buf][row][o8]);
        }
    };
    // W tile 32x128 f32 -> bf16 transposed into Bt[n][k].
    auto stageW = [&](int buf, int kk) {
        const int k = tid >> 3, c0 = (tid & 7) * 16;
        const float* src = W + (size_t)(kk + k) * N + colBase + c0;
        #pragma unroll
        for (int j = 0; j < 16; ++j) Bt[buf][c0 + j][k] = (bf16)src[j];
    };

    stageA(0, 0);
    stageW(0, 0);

    for (int kk = 0; kk < K; kk += 32) {
        const int cur = (kk >> 5) & 1;
        wait_async_copies();
        __syncthreads();                            // buffer 'cur' published
        if (kk + 32 < K) {                          // prefetch next tile
            stageA(cur ^ 1, kk + 32);
            stageW(cur ^ 1, kk + 32);
        }

        V16 afrag[4];
        #pragma unroll
        for (int mt = 0; mt < 4; ++mt)
            afrag[mt].v = *(const v16bf*)&At[cur][wm * 64 + mt * 16 + (lane & 15)][(lane >> 4) * 16];
        #pragma unroll
        for (int nt = 0; nt < 2; ++nt) {
            V16 bfrag;
            bfrag.v = *(const v16bf*)&Bt[cur][wn * 32 + nt * 16 + (lane & 15)][(lane >> 4) * 16];
            #pragma unroll
            for (int mt = 0; mt < 4; ++mt)
                acc[mt][nt].v = wmma_bf16(afrag[mt].v, bfrag.v, acc[mt][nt].v);
        }
    }

    // Epilogue: C layout is M = r + 8*(lane>=16), N = lane%16 per accum tile.
    #pragma unroll
    for (int nt = 0; nt < 2; ++nt) {
        const int col = colBase + wn * 32 + nt * 16 + (lane & 15);
        const float bval = bias[col];
        #pragma unroll
        for (int mt = 0; mt < 4; ++mt) {
            #pragma unroll
            for (int r = 0; r < 8; ++r) {
                const int row = rowBase + wm * 64 + mt * 16 + r + ((lane >> 4) << 3);
                Cout[(size_t)row * N + col] = (TOut)(acc[mt][nt].e[r] + bval);
            }
        }
    }
}

// ---------------------------------------------------------------------------
// Attention kernel: one block per (batch b, query chunk n, 64-query slab).
// 128 threads = 4 waves; each wave owns 16 query rows x all 256 keys.
// Phase 1: per key chunk m, WMMA-accumulate scores over E with async
//          double-buffered K tiles, in-register row softmax, accumulate
//          weighted probs into bf16 Psum in LDS.
// Phase 2: O = Psum @ V_n via WMMA -> bf16 "combined".
// ---------------------------------------------------------------------------
__global__ __launch_bounds__(128)
void attn_kernel(const bf16* __restrict__ Q, const bf16* __restrict__ Kv,
                 const bf16* __restrict__ Vv, bf16* __restrict__ Out) {
    __shared__ __align__(32) bf16 Psum[64][256];    // 32 KB prob-sum
    __shared__ __align__(32) bf16 Kt[2][256][32];   // 32 KB K e-step tiles
    bf16 (*Vt)[256] = reinterpret_cast<bf16 (*)[256]>(&Kt[0][0][0]);  // 8 KB alias

    const int tid  = threadIdx.x;
    const int lane = tid & 31;
    const int wv   = tid >> 5;                 // wave 0..3
    const int bx   = blockIdx.x;
    const int b    = bx >> 6;                  // 64 blocks per batch
    const int rem  = bx & 63;
    const int n    = rem >> 2;                 // query chunk
    const int qq   = rem & 3;                  // 64-row slab within chunk
    const int qg   = n * DIM_C + qq * 64;      // query row base within batch
    const size_t batchOff = (size_t)b * DIM_S * DIM_E;
    const float scale = 0.03125f;              // 1/sqrt(1024)

    for (int i = tid; i < 64 * 256; i += 128) ((bf16*)Psum)[i] = (bf16)0.0f;

    // K tile 256x32 bf16 = 1024 16B-chunks; 8 chunks/thread, async.
    auto stageK = [&](int m, int eo, int buf) {
        #pragma unroll
        for (int i = 0; i < 8; ++i) {
            const int chunk = tid + i * 128;        // 0..1023
            const int kr = chunk >> 2, o8 = (chunk & 3) * 8;
            copy16_g2l(&Kv[batchOff + (size_t)(m * DIM_C + kr) * DIM_E + eo + o8],
                       &Kt[buf][kr][o8]);
        }
    };

    V8F sacc[16];  // 16q x 256k score tile: 16 accum tiles (128 VGPRs)

    for (int m = 0; m < DIM_NC; ++m) {
        __syncthreads();                       // all readers of Kt done
        stageK(m, 0, 0);                       // pipeline prologue
        #pragma unroll
        for (int kt = 0; kt < 16; ++kt)
            #pragma unroll
            for (int r = 0; r < 8; ++r) sacc[kt].e[r] = 0.0f;

        for (int eo = 0; eo < DIM_E; eo += 32) {
            const int cur = (eo >> 5) & 1;
            wait_async_copies();
            __syncthreads();                   // Kt[cur] published
            if (eo + 32 < DIM_E) stageK(m, eo + 32, cur ^ 1);  // overlap w/ WMMA

            V16 af;  // Q fragment straight from global (L2-resident)
            af.v = *(const v16bf*)&Q[batchOff +
                       (size_t)(qg + wv * 16 + (lane & 15)) * DIM_E + eo + ((lane >> 4) * 16)];
            #pragma unroll
            for (int kt = 0; kt < 16; ++kt) {
                V16 bfrag;
                bfrag.v = *(const v16bf*)&Kt[cur][kt * 16 + (lane & 15)][(lane >> 4) * 16];
                sacc[kt].v = wmma_bf16(af.v, bfrag.v, sacc[kt].v);
            }
        }

        // Row softmax: a row lives in VGPR r across one 16-lane group
        // (C layout: M = r + 8*(lane>=16), N = lane%16).
        const float wgt = (m == n) ? 2.0f : 1.0f;  // local path == diagonal term
        #pragma unroll
        for (int r = 0; r < 8; ++r) {
            float mx = -3.4e38f;
            #pragma unroll
            for (int kt = 0; kt < 16; ++kt) {
                sacc[kt].e[r] *= scale;
                mx = fmaxf(mx, sacc[kt].e[r]);
            }
            mx = fmaxf(mx, __shfl_xor(mx, 1, 32));
            mx = fmaxf(mx, __shfl_xor(mx, 2, 32));
            mx = fmaxf(mx, __shfl_xor(mx, 4, 32));
            mx = fmaxf(mx, __shfl_xor(mx, 8, 32));
            float sum = 0.0f;
            #pragma unroll
            for (int kt = 0; kt < 16; ++kt) {
                const float p = __expf(sacc[kt].e[r] - mx);
                sacc[kt].e[r] = p;
                sum += p;
            }
            sum += __shfl_xor(sum, 1, 32);
            sum += __shfl_xor(sum, 2, 32);
            sum += __shfl_xor(sum, 4, 32);
            sum += __shfl_xor(sum, 8, 32);
            const float fac = wgt / sum;
            const int qrow = wv * 16 + r + ((lane >> 4) << 3);
            #pragma unroll
            for (int kt = 0; kt < 16; ++kt) {  // exclusive (row,col) ownership
                const int col = kt * 16 + (lane & 15);
                Psum[qrow][col] = (bf16)((float)Psum[qrow][col] + sacc[kt].e[r] * fac);
            }
        }
    }
    __syncthreads();

    // Phase 2: O = Psum @ V_n, 16-wide E tiles, V staged transposed in LDS.
    for (int et = 0; et < 64; ++et) {
        #pragma unroll
        for (int i = 0; i < 2; ++i) {          // 2 V-rows per thread (transpose)
            const int kr = tid * 2 + i;
            V16 vr;
            vr.v = *(const v16bf*)&Vv[batchOff + (size_t)(n * DIM_C + kr) * DIM_E + et * 16];
            #pragma unroll
            for (int j = 0; j < 16; ++j) Vt[j][kr] = vr.e[j];
        }
        __syncthreads();

        V8F o;
        #pragma unroll
        for (int r = 0; r < 8; ++r) o.e[r] = 0.0f;
        #pragma unroll
        for (int kk = 0; kk < 8; ++kk) {       // contract over 256 keys
            V16 af, bfrag;
            af.v    = *(const v16bf*)&Psum[wv * 16 + (lane & 15)][kk * 32 + ((lane >> 4) * 16)];
            bfrag.v = *(const v16bf*)&Vt[lane & 15][kk * 32 + ((lane >> 4) * 16)];
            o.v = wmma_bf16(af.v, bfrag.v, o.v);
        }
        #pragma unroll
        for (int r = 0; r < 8; ++r) {
            const int qrow = qg + wv * 16 + r + ((lane >> 4) << 3);
            Out[batchOff + (size_t)qrow * DIM_E + et * 16 + (lane & 15)] = (bf16)o.e[r];
        }
        __syncthreads();                       // Vt reused next iteration
    }
}

// ---------------------------------------------------------------------------
// Host launcher. Workspace (bf16): xb, Q, K, V = 4 * 32 MB = 128 MB;
// "combined" aliases xb (x is dead after the QKV projections).
// ---------------------------------------------------------------------------
extern "C" void kernel_launch(void* const* d_in, const int* in_sizes, int n_in,
                              void* d_out, int out_size, void* d_ws, size_t ws_size,
                              hipStream_t stream) {
    (void)in_sizes; (void)n_in; (void)out_size; (void)ws_size;
    const float* x  = (const float*)d_in[0];
    const float* Wq = (const float*)d_in[1];
    const float* bq = (const float*)d_in[2];
    const float* Wk = (const float*)d_in[3];
    const float* bk = (const float*)d_in[4];
    const float* Wv = (const float*)d_in[5];
    const float* bv = (const float*)d_in[6];
    const float* Wo = (const float*)d_in[7];
    const float* bo = (const float*)d_in[8];

    const size_t per = (size_t)DIM_B * DIM_S * DIM_E;  // 16 M elements
    bf16* xb  = (bf16*)d_ws;
    bf16* Qb  = xb + per;
    bf16* Kb  = Qb + per;
    bf16* Vb  = Kb + per;
    bf16* Cmb = xb;                            // alias: x dead after QKV

    const int M = DIM_B * DIM_S, N = DIM_E, K = DIM_E;
    dim3 ggrid(N / 128, M / 128);              // (8, 128)

    cvt_f32_bf16_kernel<<<per / (256 * 8), 256, 0, stream>>>(x, xb);

    gemm_bias_kernel<bf16><<<ggrid, 256, 0, stream>>>(xb, Wq, bq, Qb, M, N, K);
    gemm_bias_kernel<bf16><<<ggrid, 256, 0, stream>>>(xb, Wk, bk, Kb, M, N, K);
    gemm_bias_kernel<bf16><<<ggrid, 256, 0, stream>>>(xb, Wv, bv, Vb, M, N, K);

    attn_kernel<<<DIM_B * DIM_NC * 4, 128, 0, stream>>>(Qb, Kb, Vb, Cmb);

    gemm_bias_kernel<float><<<ggrid, 256, 0, stream>>>(Cmb, Wo, bo, (float*)d_out, M, N, K);
}
